// TemporalAttentionLayer2_58737972740365
// MI455X (gfx1250) — matrix operand order repaired and checked by reading, not compiled, not run on hardware
//
#include <hip/hip_runtime.h>

// ---------------------------------------------------------------------------
// Fused temporal attention for MI455X (gfx1250), wave32 + v_wmma bf16 path.
// kernel 1: pack f32 weights -> bf16 pre-swizzled B-fragments in d_ws.
// kernel 2: one workgroup (256 threads = 8 waves) per sequence:
//   x = inp + pos (bf16 LDS) -> Q,K,V (WMMA) -> per-head causal softmax
//   attention (WMMA) -> FF GEMM (WMMA) -> fused epilogue to HBM.
// USE_PK is a template parameter so the hot path has no fallback branches.
// ---------------------------------------------------------------------------

typedef __attribute__((ext_vector_type(16))) __bf16 v16bf;
typedef __attribute__((ext_vector_type(8)))  float  v8f;

#define TT 64     // time steps
#define FTR 256   // features
#define NH 8      // heads
#define XS 264    // padded LDS stride for 64x256 bf16 arrays (16B aligned, bank-rotating)
#define VTS 72    // padded LDS stride for transposed V (256 x 64)
#define SS 65     // padded stride for f32 scores (conflict-free softmax)
#define AS 64     // stride for bf16 attn probs

#define WFRAGS 65536               // bf16 elems per packed weight: 16 ntiles * 8 ksteps * 32 lanes * 16
#define PK_BYTES (4 * WFRAGS * 2)  // 512 KB

#define WMMA_BF16(a, b, c) \
  __builtin_amdgcn_wmma_f32_16x16x32_bf16(false, (a), false, (b), (short)0, (c), false, false)

// A fragment (16x32, MxK) from row-major bf16 LDS.
// lane<16: row m0+lane, K = k0..k0+15; lane>=16: same rows, K k0+16..k0+31.
__device__ __forceinline__ v16bf load_a16(const __bf16* p, int stride, int m0, int k0, int lane) {
  const __bf16* r = p + (m0 + (lane & 15)) * stride + k0 + ((lane & 16) ? 16 : 0);
  v16bf a;
#pragma unroll
  for (int i = 0; i < 16; ++i) a[i] = r[i];
  return a;
}

// B fragment (32x16, KxN) from a source stored as Bt[n][k] (each B-column contiguous), bf16 LDS.
// lane<16: col n0+lane holds K k0..k0+15; lane>=16: K k0+16..k0+31  (contiguous 32B per lane).
__device__ __forceinline__ v16bf load_b16_t(const __bf16* p, int stride, int n0, int k0, int lane) {
  const __bf16* r = p + (n0 + (lane & 15)) * stride + k0 + ((lane & 16) ? 16 : 0);
  v16bf b;
#pragma unroll
  for (int i = 0; i < 16; ++i) b[i] = r[i];
  return b;
}

// B fragment from pre-swizzled packed weights: 32 contiguous bytes per lane, coalesced b128s.
__device__ __forceinline__ v16bf load_bpk(const __bf16* pk, int frag, int lane) {
  const __bf16* r = pk + (((size_t)frag * 32 + lane) << 4);
  v16bf b;
#pragma unroll
  for (int i = 0; i < 16; ++i) b[i] = r[i];
  return b;
}

// Fallback B fragment from K-major f32 global weight W[k][n] (stride 256).
__device__ __forceinline__ v16bf load_b32_k(const float* W, int k0, int n0, int lane) {
  const float* c = W + (size_t)(k0 + ((lane & 16) ? 16 : 0)) * FTR + n0 + (lane & 15);
  v16bf b;
#pragma unroll
  for (int i = 0; i < 16; ++i) b[i] = (__bf16)c[i * FTR];
  return b;
}

// Fallback B fragment from f32 global stored as Bt[n][k] row-major (lin_w rows).
__device__ __forceinline__ v16bf load_b32_t(const float* W, int n0, int k0, int lane) {
  const float* r = W + (size_t)(n0 + (lane & 15)) * FTR + k0 + ((lane & 16) ? 16 : 0);
  v16bf b;
#pragma unroll
  for (int i = 0; i < 16; ++i) b[i] = (__bf16)r[i];
  return b;
}

// ---------------------------------------------------------------------------
// Kernel 1: pack Wq/Wk/Wv (K-major) and lin_w (Bt layout) into bf16 fragments.
// pk[w][frag=j*8+ks][lane][i]; total 256K elements, 1024 blocks x 256 threads.
// ---------------------------------------------------------------------------
__global__ __launch_bounds__(256)
void pack_weights(const float* __restrict__ Wq, const float* __restrict__ Wk,
                  const float* __restrict__ Wv, const float* __restrict__ LW,
                  __bf16* __restrict__ pk) {
  const int idx = blockIdx.x * 256 + threadIdx.x;
  if (idx >= 4 * WFRAGS) return;
  const int w    = idx >> 16;          // 0:Wq 1:Wk 2:Wv 3:lin_w
  const int rem  = idx & (WFRAGS - 1);
  const int frag = rem >> 9;           // j*8 + ks
  const int lane = (rem >> 4) & 31;
  const int i    = rem & 15;
  const int j  = frag >> 3, ks = frag & 7;
  const int k  = ks * 32 + ((lane & 16) ? 16 : 0) + i;
  const int nc = j * 16 + (lane & 15);
  float v;
  if (w == 3) {
    v = LW[(size_t)nc * FTR + k];                 // B[k][n] = lin_w[n][k]
  } else {
    const float* W = (w == 0) ? Wq : (w == 1) ? Wk : Wv;
    v = W[(size_t)k * FTR + nc];                  // B[k][n] = W[k][n]
  }
  pk[idx] = (__bf16)v;
}

// ---------------------------------------------------------------------------
// Kernel 2: fused attention layer, one workgroup per sequence.
// ---------------------------------------------------------------------------
template <bool USE_PK>
__global__ __launch_bounds__(256)
void fused_temporal_attention(const float* __restrict__ inp, const float* __restrict__ pos,
                              const float* __restrict__ Wq,  const float* __restrict__ Wk,
                              const float* __restrict__ Wv,  const float* __restrict__ LW,
                              const float* __restrict__ Lb,  const __bf16* __restrict__ pk,
                              float* __restrict__ out) {
  extern __shared__ char smem_raw[];
  __bf16* xh = (__bf16*)smem_raw;        // [64][XS]  x (bf16); later reused as attention output
  __bf16* Qs = xh + TT * XS;             // [64][XS]
  __bf16* Ks = Qs + TT * XS;             // [64][XS]
  __bf16* Vt = Ks + TT * XS;             // [256][VTS]  V transposed: Vt[col][row]
  float*  sc = (float*)(Vt + FTR * VTS); // [64][SS]  f32 scores
  __bf16* at = (__bf16*)(sc + TT * SS);  // [64][AS]  bf16 probs
  __bf16* oh = xh;                       // alias: attention output (xh dead after phase 1)

  const int n    = blockIdx.x;
  const int tid  = threadIdx.x;
  const int wave = tid >> 5;
  const int lane = tid & 31;
  const int mrow = (lane & 16) ? 8 : 0;  // C/D row offset for high lane half
  const int ncol = lane & 15;            // C/D column within tile

  const float* xin = inp + (size_t)n * TT * FTR;

  // ---- Phase 0: x = inputs + pos_emb -> bf16 LDS ----
  for (int idx = tid; idx < TT * FTR; idx += 256) {
    int r = idx >> 8, c = idx & 255;
    xh[r * XS + c] = (__bf16)(xin[idx] + pos[idx]);
  }
  __syncthreads();

  // ---- Phase 1: Q/K/V = x @ W  (M=64, N=256, K=256; 512 WMMA per projection) ----
  for (int task = wave; task < 48; task += 8) {
    const int p = task >> 4;   // 0:Q 1:K 2:V
    const int j = task & 15;   // n-tile (16 cols)
    v8f acc[4] = {};
#pragma unroll
    for (int ks = 0; ks < 8; ++ks) {
      v16bf b;
      if (USE_PK) {
        b = load_bpk(pk + (size_t)p * WFRAGS, j * 8 + ks, lane);
      } else {
        const float* W = (p == 0) ? Wq : (p == 1) ? Wk : Wv;
        b = load_b32_k(W, ks * 32, j * 16, lane);
      }
#pragma unroll
      for (int m = 0; m < 4; ++m) {
        v16bf a = load_a16(xh, XS, m * 16, ks * 32, lane);
        acc[m] = WMMA_BF16(a, b, acc[m]);
      }
    }
    if (p == 2) {  // V stored transposed for contiguous P@V B-fragments
#pragma unroll
      for (int m = 0; m < 4; ++m)
#pragma unroll
        for (int i = 0; i < 8; ++i)
          Vt[(j * 16 + ncol) * VTS + (m * 16 + mrow + i)] = (__bf16)acc[m][i];
    } else {
      __bf16* D = p ? Ks : Qs;
#pragma unroll
      for (int m = 0; m < 4; ++m)
#pragma unroll
        for (int i = 0; i < 8; ++i)
          D[(m * 16 + mrow + i) * XS + j * 16 + ncol] = (__bf16)acc[m][i];
    }
  }

  // Prefetch this block's input (64KB) for the fused epilogue re-read; the whole
  // attention phase covers the latency. Emits gfx1250 global_prefetch.
  for (int idx = tid * 16; idx < TT * FTR; idx += 256 * 16)
    __builtin_prefetch(xin + idx, 0, 3);

  __syncthreads();

  // ---- Phase 2: per-head causal attention ----
  const float inv_scale = 0.125f;  // 1/sqrt(NUM_TIME_STEPS)
  for (int h = 0; h < NH; ++h) {
    // scores = Qh @ Kh^T  (16 tiles over 8 waves)
    for (int t2 = wave; t2 < 16; t2 += 8) {
      const int mi = t2 >> 2, ni = t2 & 3;
      v8f acc = {};
      v16bf a = load_a16(Qs + h * 32, XS, mi * 16, 0, lane);
      v16bf b = load_b16_t(Ks + h * 32, XS, ni * 16, 0, lane);  // B col n = K row n
      acc = WMMA_BF16(a, b, acc);
#pragma unroll
      for (int i = 0; i < 8; ++i)
        sc[(mi * 16 + mrow + i) * SS + ni * 16 + ncol] = acc[i] * inv_scale;
    }
    __syncthreads();

    // causal softmax: one thread per row (cols > row get prob 0, matching exp(NEG_INF) = 0)
    if (tid < TT) {
      const int r = tid;
      float* srow = sc + r * SS;
      float mx = srow[0];
      for (int c = 1; c <= r; ++c) mx = fmaxf(mx, srow[c]);
      float sum = 0.f;
      for (int c = 0; c <= r; ++c) { float e = __expf(srow[c] - mx); srow[c] = e; sum += e; }
      const float inv = 1.0f / sum;
      __bf16* arow = at + r * AS;
      for (int c = 0; c < TT; ++c) arow[c] = (__bf16)((c <= r) ? srow[c] * inv : 0.0f);
    }
    __syncthreads();

    // out_h = P @ Vh  (8 tiles, one per wave; K=64 -> 2 WMMA steps)
    {
      const int mi = wave >> 1, ni = wave & 1;
      v8f acc = {};
#pragma unroll
      for (int ks = 0; ks < 2; ++ks) {
        v16bf a = load_a16(at, AS, mi * 16, ks * 32, lane);
        v16bf b = load_b16_t(Vt, VTS, h * 32 + ni * 16, ks * 32, lane);  // contiguous rows of Vt
        acc = WMMA_BF16(a, b, acc);
      }
#pragma unroll
      for (int i = 0; i < 8; ++i)
        oh[(mi * 16 + mrow + i) * XS + h * 32 + ni * 16 + ncol] = (__bf16)acc[i];
    }
    __syncthreads();
  }

  // ---- Phase 3: FF GEMM + fused epilogue:  relu(out@lin_w^T + b) + out + x -> HBM ----
  for (int task = wave; task < 64; task += 8) {
    const int mi = task >> 4, nj = task & 15;
    v8f acc = {};
#pragma unroll
    for (int ks = 0; ks < 8; ++ks) {
      v16bf a = load_a16(oh, XS, mi * 16, ks * 32, lane);
      v16bf b;
      if (USE_PK) b = load_bpk(pk + (size_t)3 * WFRAGS, nj * 8 + ks, lane);
      else        b = load_b32_t(LW, nj * 16, ks * 32, lane);  // B[k][n] = lin_w[n][k]
      acc = WMMA_BF16(a, b, acc);
    }
#pragma unroll
    for (int i = 0; i < 8; ++i) {
      const int row = mi * 16 + mrow + i;
      const int col = nj * 16 + ncol;
      const float o  = (float)oh[row * XS + col];
      const float ff = fmaxf(acc[i] + Lb[col], 0.0f) + o;
      const float xv = xin[row * FTR + col] + pos[row * FTR + col];  // exact f32 residual
      out[(size_t)n * TT * FTR + row * FTR + col] = ff + xv;
    }
  }
}

// LDS: 3*64*264*2 (x/Q/K) + 256*72*2 (Vt) + 64*65*4 (sc) + 64*64*2 (at) = 163072 bytes
#define SMEM_BYTES 163072

extern "C" void kernel_launch(void* const* d_in, const int* in_sizes, int n_in,
                              void* d_out, int out_size, void* d_ws, size_t ws_size,
                              hipStream_t stream) {
  (void)in_sizes; (void)n_in; (void)out_size;
  const float* inp = (const float*)d_in[0];
  const float* pos = (const float*)d_in[1];
  const float* Wq  = (const float*)d_in[2];
  const float* Wk  = (const float*)d_in[3];
  const float* Wv  = (const float*)d_in[4];
  const float* LW  = (const float*)d_in[5];
  const float* Lb  = (const float*)d_in[6];
  float* o = (float*)d_out;

  const bool use_pk = (d_ws != nullptr && ws_size >= (size_t)PK_BYTES);
  __bf16* pk = (__bf16*)d_ws;

  const int N = 4096;  // sequences; one workgroup each
  if (use_pk) {
    pack_weights<<<(4 * WFRAGS) / 256, 256, 0, stream>>>(Wq, Wk, Wv, LW, pk);
    hipFuncSetAttribute((const void*)fused_temporal_attention<true>,
                        hipFuncAttributeMaxDynamicSharedMemorySize, SMEM_BYTES);
    fused_temporal_attention<true><<<N, 256, SMEM_BYTES, stream>>>(inp, pos, Wq, Wk, Wv, LW, Lb,
                                                                   pk, o);
  } else {
    hipFuncSetAttribute((const void*)fused_temporal_attention<false>,
                        hipFuncAttributeMaxDynamicSharedMemorySize, SMEM_BYTES);
    fused_temporal_attention<false><<<N, 256, SMEM_BYTES, stream>>>(inp, pos, Wq, Wk, Wv, LW, Lb,
                                                                    pk, o);
  }
}
